// MultiHeadAttention_1486058684552
// MI455X (gfx1250) — matrix operand order repaired and checked by reading, not compiled, not run on hardware
//
#include <hip/hip_runtime.h>
#include <hip/hip_bf16.h>

// ---------------- problem constants ----------------
constexpr int B_  = 4;
constexpr int S_  = 2048;
constexpr int DM_ = 1024;
constexpr int H_  = 16;
constexpr int DH_ = 64;        // head depth
constexpr int M_  = B_ * S_;   // 8192 rows

typedef __attribute__((ext_vector_type(16))) __bf16 v16bf;
typedef __attribute__((ext_vector_type(8)))  float  v8f;

union FragU { uint4 q[2]; v16bf v; };

static __device__ __forceinline__ v8f vzero8() {
  v8f z = {0.f,0.f,0.f,0.f,0.f,0.f,0.f,0.f};
  return z;
}

static __device__ __forceinline__ v8f wmma_bf16(v16bf a, v16bf b, v8f c) {
  return __builtin_amdgcn_wmma_f32_16x16x32_bf16(
      false, a, false, b, (short)0, c, false, false);
}

// A fragment: 16x32 bf16, row m = lane&15, half = lane>>4.
// VGPR0..3 <- K = half*8 .. +7 ; VGPR4..7 <- K = 16+half*8 .. +7
static __device__ __forceinline__ v16bf load_frag_a(const __bf16* base, int strideElems) {
  const int lane = threadIdx.x & 31;
  const int m = lane & 15, half = lane >> 4;
  const char* p = (const char*)(base + (size_t)m * strideElems + half * 8);
  FragU f;
  f.q[0] = *(const uint4*)(p);
  f.q[1] = *(const uint4*)(p + 32);
  return f.v;
}

// B fragment: 32x16 bf16, column n = lane&15, half = lane>>4.
// VGPR0..3 <- K = half*8 .. +7 ; VGPR4..7 <- K = 16+half*8 .. +7
static __device__ __forceinline__ v16bf load_frag_b(const __bf16* base, int strideElems) {
  const int lane = threadIdx.x & 31;
  const int n = lane & 15, half = lane >> 4;
  const char* p = (const char*)(base + (size_t)n * strideElems + half * 8);
  FragU f;
  f.q[0] = *(const uint4*)(p);
  f.q[1] = *(const uint4*)(p + 32);
  return f.v;
}

static __device__ __forceinline__ void cvt_store8(__bf16* dst, float4 a, float4 b) {
  union { __bf16 h[8]; uint4 q; } u;
  u.h[0] = (__bf16)a.x; u.h[1] = (__bf16)a.y; u.h[2] = (__bf16)a.z; u.h[3] = (__bf16)a.w;
  u.h[4] = (__bf16)b.x; u.h[5] = (__bf16)b.y; u.h[6] = (__bf16)b.z; u.h[7] = (__bf16)b.w;
  *(uint4*)dst = u.q;
}

// CDNA5 async copy: 16 bytes global -> LDS, tracked by ASYNCcnt (§15.18 op 98).
static __device__ __forceinline__ void async_copy16(void* lds, const void* gsrc) {
  asm volatile("global_load_async_to_lds_b128 %0, %1, off"
               :: "v"((unsigned)(size_t)lds), "v"(gsrc)
               : "memory");
}
static __device__ __forceinline__ void async_wait0() {
  asm volatile("s_wait_asynccnt 0x0" ::: "memory");
}

// =====================================================================
// One-time precision/layout conversion kernels.
// =====================================================================
__global__ __launch_bounds__(256) void k_cvt_x(const float* __restrict__ X,
                                               __bf16* __restrict__ Xb) {
  const size_t i = ((size_t)blockIdx.x * 256 + threadIdx.x) * 8;
  float4 a = *(const float4*)(X + i);
  float4 b = *(const float4*)(X + i + 4);
  cvt_store8(Xb + i, a, b);
}

// fp32 W[K][N] -> bf16 Wt[N][K] (convert + transpose, 32x32 tiles via LDS)
__global__ __launch_bounds__(256) void k_cvt_w(const float* __restrict__ W,
                                               __bf16* __restrict__ Wt) {
  __shared__ float sT[32][33];
  const int t  = threadIdx.x;
  const int k0 = blockIdx.y * 32, n0 = blockIdx.x * 32;
  {
    const int r = t >> 3, c = (t & 7) * 4;
    const float4 f = *(const float4*)(W + (size_t)(k0 + r) * DM_ + n0 + c);
    sT[r][c] = f.x; sT[r][c + 1] = f.y; sT[r][c + 2] = f.z; sT[r][c + 3] = f.w;
  }
  __syncthreads();
  {
    const int rn = t >> 3, ck = (t & 7) * 4;
    union { __bf16 h[4]; uint2 q; } u;
    u.h[0] = (__bf16)sT[ck + 0][rn]; u.h[1] = (__bf16)sT[ck + 1][rn];
    u.h[2] = (__bf16)sT[ck + 2][rn]; u.h[3] = (__bf16)sT[ck + 3][rn];
    *(uint2*)(Wt + (size_t)(n0 + rn) * DM_ + k0 + ck) = u.q;
  }
}

// =====================================================================
// Y = X(bf16)[M,K] @ Wt(bf16 [N][K])^T + bias, tile 128x128x32.
// Double-buffered LDS; tiles staged with GLOBAL_LOAD_ASYNC_TO_LDS_B128 so
// the copy of step i+1 overlaps the WMMAs of step i.
// TOUT=false: Y[m][n] ; TOUT=true: Y[n][m] (for V projection).
// =====================================================================
template <bool TOUT>
__global__ __launch_bounds__(256) void k_proj(const __bf16* __restrict__ X,
                                              const __bf16* __restrict__ Wt,
                                              const float* __restrict__ bias,
                                              __bf16* __restrict__ Y) {
  __shared__ __bf16 sA[2][128 * 40];
  __shared__ __bf16 sB[2][128 * 40];

  const int t    = threadIdx.x;
  const int wave = t >> 5;
  const int lane = t & 31;
  const int wm   = wave >> 1;   // 0..3
  const int wn   = wave & 1;    // 0..1
  const int bm   = blockIdx.y, bn = blockIdx.x;

  v8f acc[2][4];
#pragma unroll
  for (int i = 0; i < 2; ++i)
#pragma unroll
    for (int j = 0; j < 4; ++j) acc[i][j] = vzero8();

  const int rA = t >> 1, cA = (t & 1) * 16;   // 128 x 32 bf16, 16 elems/thread
  const __bf16* srcA0 = X  + (size_t)(bm * 128 + rA) * DM_ + cA;
  const __bf16* srcB0 = Wt + (size_t)(bn * 128 + rA) * DM_ + cA;

  auto issue = [&](int buf, int kk) {
    __bf16* dA = &sA[buf][rA * 40 + cA];
    __bf16* dB = &sB[buf][rA * 40 + cA];
    async_copy16(dA,     srcA0 + kk);
    async_copy16(dA + 8, srcA0 + kk + 8);
    async_copy16(dB,     srcB0 + kk);
    async_copy16(dB + 8, srcB0 + kk + 8);
  };

  issue(0, 0);
  async_wait0();
  __syncthreads();

  int cur = 0;
  for (int kk = 0; kk < DM_; kk += 32) {
    const bool more = (kk + 32 < DM_);
    if (more) issue(cur ^ 1, kk + 32);

    v16bf afr[2], bfr[4];
#pragma unroll
    for (int i = 0; i < 2; ++i)
      afr[i] = load_frag_a(&sA[cur][(wm * 32 + i * 16) * 40], 40);
#pragma unroll
    for (int j = 0; j < 4; ++j)
      bfr[j] = load_frag_b(&sB[cur][(wn * 64 + j * 16) * 40], 40);
#pragma unroll
    for (int i = 0; i < 2; ++i)
#pragma unroll
      for (int j = 0; j < 4; ++j) acc[i][j] = wmma_bf16(afr[i], bfr[j], acc[i][j]);

    if (more) {
      async_wait0();
      __syncthreads();
      cur ^= 1;
    }
  }

  const int lm = lane & 15, hf = lane >> 4;
#pragma unroll
  for (int i = 0; i < 2; ++i)
#pragma unroll
    for (int j = 0; j < 4; ++j) {
      const int gn  = bn * 128 + wn * 64 + j * 16 + lm;
      const int gm0 = bm * 128 + wm * 32 + i * 16 + hf * 8;
      const float bv = bias[gn];
      if (TOUT) {   // Y[n][m]: 8 contiguous elems -> one 16B store
        union { __bf16 h[8]; uint4 q; } u;
#pragma unroll
        for (int r = 0; r < 8; ++r) u.h[r] = (__bf16)(acc[i][j][r] + bv);
        *(uint4*)(Y + (size_t)gn * M_ + gm0) = u.q;
      } else {
#pragma unroll
        for (int r = 0; r < 8; ++r)
          Y[(size_t)(gm0 + r) * DM_ + gn] = (__bf16)(acc[i][j][r] + bv);
      }
    }
}

// =====================================================================
// scores = softmax((q @ k^T)/8 + mask*-1e9), fp32 weights out.
// One block per (b,h,16 query rows); accumulators for all 2048 keys in regs.
// =====================================================================
__global__ __launch_bounds__(256) void k_scores_softmax(
    const __bf16* __restrict__ qp, const __bf16* __restrict__ kp,
    const float* __restrict__ mask, float* __restrict__ wout) {
  __shared__ __bf16 sQ[16 * 72];
  __shared__ float  sred[8 * 16];
  __shared__ float  srow[16];

  const int blk = blockIdx.x;
  const int qt  = blk & 127;
  const int bh  = blk >> 7;
  const int h   = bh & 15;
  const int b   = bh >> 4;
  const int s0  = qt * 16;

  const int t    = threadIdx.x;
  const int wave = t >> 5;
  const int lane = t & 31;

  {
    const int r = t >> 4, c = (t & 15) * 4;
    const __bf16* src = qp + ((size_t)b * S_ + s0 + r) * DM_ + h * DH_ + c;
    *(uint2*)&sQ[r * 72 + c] = *(const uint2*)src;
  }
  __syncthreads();

  const v16bf aq0 = load_frag_a(sQ, 72);
  const v16bf aq1 = load_frag_a(sQ + 32, 72);

  const int   colBase = wave * 256;
  const float scale   = 0.125f;
  v8f acc[16];

#pragma unroll
  for (int tt = 0; tt < 16; ++tt) {
    const int key0 = colBase + tt * 16;
    const __bf16* kbase = kp + ((size_t)b * S_ + key0) * DM_ + h * DH_;
    v16bf b0 = load_frag_b(kbase, DM_);
    v16bf b1 = load_frag_b(kbase + 32, DM_);
    v8f c = vzero8();
    c = wmma_bf16(aq0, b0, c);
    c = wmma_bf16(aq1, b1, c);
    const float mv = -1e9f * mask[b * S_ + key0 + (lane & 15)];
#pragma unroll
    for (int r = 0; r < 8; ++r) acc[tt][r] = c[r] * scale + mv;
  }

  const int hf = lane >> 4;
  float pst[8];
#pragma unroll
  for (int r = 0; r < 8; ++r) {
    float m = -3.0e38f;
#pragma unroll
    for (int tt = 0; tt < 16; ++tt) m = fmaxf(m, acc[tt][r]);
    pst[r] = m;
  }
#pragma unroll
  for (int xm = 8; xm >= 1; xm >>= 1)
#pragma unroll
    for (int r = 0; r < 8; ++r) pst[r] = fmaxf(pst[r], __shfl_xor(pst[r], xm, 32));
  if ((lane & 15) == 0) {
#pragma unroll
    for (int r = 0; r < 8; ++r) sred[wave * 16 + hf * 8 + r] = pst[r];
  }
  __syncthreads();
  if (t < 16) {
    float m = -3.0e38f;
    for (int w = 0; w < 8; ++w) m = fmaxf(m, sred[w * 16 + t]);
    srow[t] = m;
  }
  __syncthreads();

  float rstat[8];
#pragma unroll
  for (int r = 0; r < 8; ++r) rstat[r] = srow[hf * 8 + r];

  float psum[8];
#pragma unroll
  for (int r = 0; r < 8; ++r) psum[r] = 0.f;
#pragma unroll
  for (int tt = 0; tt < 16; ++tt)
#pragma unroll
    for (int r = 0; r < 8; ++r) {
      float e = __expf(acc[tt][r] - rstat[r]);
      acc[tt][r] = e;
      psum[r] += e;
    }
#pragma unroll
  for (int xm = 8; xm >= 1; xm >>= 1)
#pragma unroll
    for (int r = 0; r < 8; ++r) psum[r] += __shfl_xor(psum[r], xm, 32);
  if ((lane & 15) == 0) {
#pragma unroll
    for (int r = 0; r < 8; ++r) sred[wave * 16 + hf * 8 + r] = psum[r];
  }
  __syncthreads();
  if (t < 16) {
    float s = 0.f;
    for (int w = 0; w < 8; ++w) s += sred[w * 16 + t];
    srow[t] = s;
  }
  __syncthreads();
#pragma unroll
  for (int r = 0; r < 8; ++r) rstat[r] = 1.0f / srow[hf * 8 + r];

  float* obase = wout + (((size_t)(b * H_ + h)) * S_ + s0) * S_;
#pragma unroll
  for (int tt = 0; tt < 16; ++tt) {
    const int col = colBase + tt * 16 + (lane & 15);
#pragma unroll
    for (int r = 0; r < 8; ++r)
      obase[(size_t)(hf * 8 + r) * S_ + col] = acc[tt][r] * rstat[r];
  }
}

// =====================================================================
// attn = weights(fp32->bf16) @ v_head ; V pre-transposed: vpT[d][m].
// Double-buffered: B tile async-copied, A tile cvt pipeline; both staged
// one k-step ahead of the WMMAs.
// =====================================================================
__global__ __launch_bounds__(256) void k_attn_v(const float* __restrict__ wts,
                                                const __bf16* __restrict__ vpT,
                                                __bf16* __restrict__ attn) {
  __shared__ __bf16 sA[2][128 * 40];   // weights tile [row][k]
  __shared__ __bf16 sB[2][64 * 40];    // v tile [d][sk]

  const int t    = threadIdx.x;
  const int wave = t >> 5;
  const int lane = t & 31;
  const int wm   = wave >> 1;
  const int wn   = wave & 1;
  const int bh   = blockIdx.y;
  const int b    = bh >> 4, h = bh & 15;
  const int m0   = blockIdx.x * 128;

  v8f acc[2][2];
#pragma unroll
  for (int i = 0; i < 2; ++i)
#pragma unroll
    for (int j = 0; j < 2; ++j) acc[i][j] = vzero8();

  const int rA = t >> 1, cA = (t & 1) * 16;   // 128 x 32 fp32
  const int dB = t >> 2, cB = (t & 3) * 8;    // 64 d-rows x 32 sk

  const float*  srcA0 = wts + (size_t)bh * S_ * S_ + (size_t)(m0 + rA) * S_ + cA;
  const __bf16* srcB0 = vpT + (size_t)(h * DH_ + dB) * M_ + b * S_ + cB;

  auto issue = [&](int buf, int kk) {
    async_copy16(&sB[buf][dB * 40 + cB], srcB0 + kk);
    const float* src = srcA0 + kk;
    float4 f0 = *(const float4*)(src + 0);
    float4 f1 = *(const float4*)(src + 4);
    float4 f2 = *(const float4*)(src + 8);
    float4 f3 = *(const float4*)(src + 12);
    cvt_store8(&sA[buf][rA * 40 + cA + 0], f0, f1);
    cvt_store8(&sA[buf][rA * 40 + cA + 8], f2, f3);
  };

  issue(0, 0);
  async_wait0();
  __syncthreads();

  int cur = 0;
  for (int kk = 0; kk < S_; kk += 32) {
    const bool more = (kk + 32 < S_);
    if (more) issue(cur ^ 1, kk + 32);

    v16bf afr[2], bfr[2];
#pragma unroll
    for (int i = 0; i < 2; ++i)
      afr[i] = load_frag_a(&sA[cur][(wm * 32 + i * 16) * 40], 40);
#pragma unroll
    for (int j = 0; j < 2; ++j)
      bfr[j] = load_frag_b(&sB[cur][(wn * 32 + j * 16) * 40], 40);
#pragma unroll
    for (int i = 0; i < 2; ++i)
#pragma unroll
      for (int j = 0; j < 2; ++j) acc[i][j] = wmma_bf16(afr[i], bfr[j], acc[i][j]);

    if (more) {
      async_wait0();
      __syncthreads();
      cur ^= 1;
    }
  }

  const int lm = lane & 15, hf = lane >> 4;
#pragma unroll
  for (int i = 0; i < 2; ++i)
#pragma unroll
    for (int j = 0; j < 2; ++j) {
      const int gm = m0 + wm * 32 + i * 16 + hf * 8;
      const int gd = wn * 32 + j * 16 + lm;
#pragma unroll
      for (int r = 0; r < 8; ++r)
        attn[((size_t)bh * S_ + gm + r) * DH_ + gd] = (__bf16)(acc[i][j][r]);
    }
}

// =====================================================================
// out = concat(attn) @ WoT^T + bo, fp32 out. Double-buffered, async tiles.
// =====================================================================
__global__ __launch_bounds__(256) void k_out_proj(const __bf16* __restrict__ attn,
                                                  const __bf16* __restrict__ WoT,
                                                  const float* __restrict__ bias,
                                                  float* __restrict__ out) {
  __shared__ __bf16 sA[2][128 * 40];
  __shared__ __bf16 sB[2][128 * 40];

  const int t    = threadIdx.x;
  const int wave = t >> 5;
  const int lane = t & 31;
  const int wm   = wave >> 1;
  const int wn   = wave & 1;
  const int bm   = blockIdx.y, bn = blockIdx.x;

  v8f acc[2][4];
#pragma unroll
  for (int i = 0; i < 2; ++i)
#pragma unroll
    for (int j = 0; j < 4; ++j) acc[i][j] = vzero8();

  const int rA = t >> 1, cA = (t & 1) * 16;
  const int gr = bm * 128 + rA;
  const int ab = gr >> 11, as = gr & (S_ - 1);

  const __bf16* attnRow = attn + ((size_t)ab * H_ * S_ + as) * DH_;   // + h*S_*DH_ later
  const __bf16* srcB0   = WoT + (size_t)(bn * 128 + rA) * DM_ + cA;

  auto issue = [&](int buf, int kk) {
    const int hh = kk >> 6, d0 = (kk & 63) + cA;   // 32-chunk stays in one head
    const __bf16* srcA = attnRow + (size_t)hh * S_ * DH_ + d0;
    __bf16* dA = &sA[buf][rA * 40 + cA];
    __bf16* dB = &sB[buf][rA * 40 + cA];
    async_copy16(dA,     srcA);
    async_copy16(dA + 8, srcA + 8);
    async_copy16(dB,     srcB0 + kk);
    async_copy16(dB + 8, srcB0 + kk + 8);
  };

  issue(0, 0);
  async_wait0();
  __syncthreads();

  int cur = 0;
  for (int kk = 0; kk < DM_; kk += 32) {
    const bool more = (kk + 32 < DM_);
    if (more) issue(cur ^ 1, kk + 32);

    v16bf afr[2], bfr[4];
#pragma unroll
    for (int i = 0; i < 2; ++i)
      afr[i] = load_frag_a(&sA[cur][(wm * 32 + i * 16) * 40], 40);
#pragma unroll
    for (int j = 0; j < 4; ++j)
      bfr[j] = load_frag_b(&sB[cur][(wn * 64 + j * 16) * 40], 40);
#pragma unroll
    for (int i = 0; i < 2; ++i)
#pragma unroll
      for (int j = 0; j < 4; ++j) acc[i][j] = wmma_bf16(afr[i], bfr[j], acc[i][j]);

    if (more) {
      async_wait0();
      __syncthreads();
      cur ^= 1;
    }
  }

  const int lm = lane & 15, hf = lane >> 4;
#pragma unroll
  for (int i = 0; i < 2; ++i)
#pragma unroll
    for (int j = 0; j < 4; ++j) {
      const int gn  = bn * 128 + wn * 64 + j * 16 + lm;
      const int gm0 = bm * 128 + wm * 32 + i * 16 + hf * 8;
      const float bv = bias[gn];
#pragma unroll
      for (int r = 0; r < 8; ++r)
        out[(size_t)(gm0 + r) * DM_ + gn] = acc[i][j][r] + bv;
    }
}

// =====================================================================
extern "C" void kernel_launch(void* const* d_in, const int* in_sizes, int n_in,
                              void* d_out, int out_size, void* d_ws, size_t ws_size,
                              hipStream_t stream) {
  const float* Q    = (const float*)d_in[0];
  const float* K    = (const float*)d_in[1];
  const float* V    = (const float*)d_in[2];
  const float* mask = (const float*)d_in[3];
  const float* Wq   = (const float*)d_in[4];
  const float* bq   = (const float*)d_in[5];
  const float* Wk   = (const float*)d_in[6];
  const float* bk   = (const float*)d_in[7];
  const float* Wv   = (const float*)d_in[8];
  const float* bv   = (const float*)d_in[9];
  const float* Wo   = (const float*)d_in[10];
  const float* bo   = (const float*)d_in[11];

  float* out = (float*)d_out;
  float* wts = out + (size_t)M_ * DM_;               // weights region of d_out

  const size_t N  = (size_t)M_ * DM_;                // 8.4M elements
  const size_t NW = (size_t)DM_ * DM_;               // 1M elements
  __bf16* xb   = (__bf16*)d_ws;                      // reused for Q,K,V bf16
  __bf16* wtq  = xb  + N;
  __bf16* wtk  = wtq + NW;
  __bf16* wtv  = wtk + NW;
  __bf16* wto  = wtv + NW;
  __bf16* qp   = wto + NW;                           // [M][DM] bf16
  __bf16* kp   = qp  + N;                            // [M][DM] bf16
  __bf16* vpT  = kp  + N;                            // [DM][M] bf16 (transposed)
  __bf16* attn = vpT + N;                            // [B,H,S,64] bf16

  dim3 gW(DM_ / 32, DM_ / 32);                       // (32,32)
  k_cvt_w<<<gW, 256, 0, stream>>>(Wq, wtq);
  k_cvt_w<<<gW, 256, 0, stream>>>(Wk, wtk);
  k_cvt_w<<<gW, 256, 0, stream>>>(Wv, wtv);
  k_cvt_w<<<gW, 256, 0, stream>>>(Wo, wto);

  dim3 gProj(DM_ / 128, M_ / 128);                   // (8, 64)
  const int gX = (int)(N / (256 * 8));               // 4096

  k_cvt_x<<<gX, 256, 0, stream>>>(Q, xb);
  k_proj<false><<<gProj, 256, 0, stream>>>(xb, wtq, bq, qp);
  k_cvt_x<<<gX, 256, 0, stream>>>(K, xb);
  k_proj<false><<<gProj, 256, 0, stream>>>(xb, wtk, bk, kp);
  k_cvt_x<<<gX, 256, 0, stream>>>(V, xb);
  k_proj<true><<<gProj, 256, 0, stream>>>(xb, wtv, bv, vpT);

  k_scores_softmax<<<dim3(B_ * H_ * (S_ / 16)), 256, 0, stream>>>(qp, kp, mask, wts);

  k_attn_v<<<dim3(S_ / 128, B_ * H_), 256, 0, stream>>>(wts, vpT, attn);

  k_out_proj<<<gProj, 256, 0, stream>>>(attn, wto, bo, out);
}
